// SimpleTransformer_72524817760447
// MI455X (gfx1250) — compile-verified, matrix operand order checked by reading, not run on hardware
//
#include <hip/hip_runtime.h>
#include <hip/hip_bf16.h>
#include <math.h>

// ---------------------------------------------------------------------------
// CDNA5 (gfx1250, wave32) transformer forward.
// All GEMMs use v_wmma_f32_16x16x32_bf16 (f32->bf16 via v_cvt_pk_bf16_f32 on
// the LDS staging path, f32 accumulate). Double-buffered LDS pipeline.
// ---------------------------------------------------------------------------

typedef __attribute__((ext_vector_type(16))) __bf16 v16bf;
typedef __attribute__((ext_vector_type(8)))  __bf16 v8bf;
typedef __attribute__((ext_vector_type(4)))  __bf16 v4bf;
typedef __attribute__((ext_vector_type(2)))  __bf16 v2bf;
typedef __attribute__((ext_vector_type(8)))  float  v8f;

__device__ __forceinline__ v2bf cvt2(float a, float b) {
#if __has_builtin(__builtin_amdgcn_cvt_pk_bf16_f32)
  return __builtin_amdgcn_cvt_pk_bf16_f32(a, b);
#else
  v2bf r;
  r.x = (__bf16)a;
  r.y = (__bf16)b;
  return r;
#endif
}

__device__ __forceinline__ v4bf cvt4(float x, float y, float z, float w) {
  const v2bf p0 = cvt2(x, y);
  const v2bf p1 = cvt2(z, w);
  return __builtin_shufflevector(p0, p1, 0, 1, 2, 3);
}

union FragU { unsigned int u[8]; v16bf v; };

// Per-lane fragment gather from LDS tile stored [row][K], row stride 40 elems.
// 16-bit A-matrix 16x32 layout (ISA 7.12.2): VGPR p holds K pair
// k0 = 2*(p&3) + 16*(p>>2) + 8*half. Offsets form two contiguous 16B chunks,
// so this lowers to 2x ds_load_b128 per fragment.
__device__ __forceinline__ v16bf lds_frag(const __bf16* base, int row, int half) {
  FragU f;
#pragma unroll
  for (int p = 0; p < 8; ++p) {
    const int k0 = 2 * (p & 3) + 16 * (p >> 2) + 8 * half;
    f.u[p] = *reinterpret_cast<const unsigned int*>(base + row * 40 + k0);
  }
  return f.v;
}

__device__ __forceinline__ v8f wmma_bf16(v16bf a, v16bf b, v8f c) {
  return __builtin_amdgcn_wmma_f32_16x16x32_bf16(
      /*neg_a=*/false, a, /*neg_b=*/false, b,
      /*c_mod=*/(short)0, c, /*reuse_a=*/false, /*reuse_b=*/false);
}

// EPI: 0 = none, 1 = +bias, 2 = +bias then exact GELU, 3 = +bias +residual.
// TRANSB: B stored [N,K] row-major (used for Q @ K^T).
// Block tile 128(M) x 128(N), K-step 32. 256 threads = 8 waves,
// each wave computes 32x64 as 2x4 WMMA 16x16 tiles. Double-buffered LDS.
template <int EPI, bool TRANSB>
__global__ __launch_bounds__(256) void gemm_bf16_wmma(
    const float* __restrict__ A, const float* __restrict__ Bm,
    const float* __restrict__ bias, const float* __restrict__ Res,
    float* __restrict__ C, int M, int N, int K,
    long long strideA, long long strideB, long long strideC) {
  __shared__ alignas(16) __bf16 sA[2][128 * 40];  // [m][k]
  __shared__ alignas(16) __bf16 sB[2][128 * 40];  // [n][k]

  const int z = blockIdx.z;
  A  += (size_t)z * (size_t)strideA;
  Bm += (size_t)z * (size_t)strideB;
  C  += (size_t)z * (size_t)strideC;
  if constexpr (EPI == 3) Res += (size_t)z * (size_t)strideC;

  const int tileN = blockIdx.x * 128;
  const int tileM = blockIdx.y * 128;
  const int t     = threadIdx.x;
  const int wid   = t >> 5;
  const int lane  = t & 31;
  const int waveM = wid & 3;   // 4 waves along M (32 rows each)
  const int waveN = wid >> 2;  // 2 waves along N (64 cols each)
  const int lm    = lane & 15;
  const int half  = lane >> 4;

  // ---- staging geometry (hoisted out of the K loop) ----
  const int rowL = t >> 3;   // 0..31: local row (A tile / B^T tile), step 32/i
  const int c4   = t & 7;    // float4 slot within a 32-wide K row
  const int nB   = t & 127;  // [K,N]-major B: one N column per thread
  const int kg   = t >> 7;   // [K,N]-major B: K half (16 rows each)

  const float* aPtr  = A  + (size_t)(tileM + rowL) * K + c4 * 4;        // +kk
  const float* bPtrT = Bm + (size_t)(tileN + rowL) * K + c4 * 4;        // +kk
  const float* bPtrN = Bm + (size_t)kg * 16 * N + tileN + nB;           // +kk*N
  const int ldsAoff  = rowL * 40 + c4 * 4;                              // 8B aligned
  const int ldsBoffN = nB * 40 + kg * 16;                               // 16B aligned

  float4 ra[4];
  float4 rbT[4];
  float  rbN[16];

  auto load_tiles = [&](int kk) {
#pragma unroll
    for (int i = 0; i < 4; ++i)
      ra[i] = *reinterpret_cast<const float4*>(aPtr + (size_t)i * 32 * K + kk);
    if constexpr (TRANSB) {
#pragma unroll
      for (int i = 0; i < 4; ++i)
        rbT[i] = *reinterpret_cast<const float4*>(bPtrT + (size_t)i * 32 * K + kk);
    } else {
      // one column, 16 consecutive K rows: lanes cover consecutive n -> coalesced
#pragma unroll
      for (int i = 0; i < 16; ++i)
        rbN[i] = bPtrN[(size_t)(kk + i) * N];
    }
  };

  auto store_tiles = [&](__bf16* dA, __bf16* dB) {
#pragma unroll
    for (int i = 0; i < 4; ++i)
      *reinterpret_cast<v4bf*>(&dA[ldsAoff + i * 32 * 40]) =
          cvt4(ra[i].x, ra[i].y, ra[i].z, ra[i].w);
    if constexpr (TRANSB) {
#pragma unroll
      for (int i = 0; i < 4; ++i)
        *reinterpret_cast<v4bf*>(&dB[ldsAoff + i * 32 * 40]) =
            cvt4(rbT[i].x, rbT[i].y, rbT[i].z, rbT[i].w);
    } else {
      // 16 bf16 along K for one n: two contiguous 16B LDS stores
      const v4bf a0 = cvt4(rbN[0],  rbN[1],  rbN[2],  rbN[3]);
      const v4bf a1 = cvt4(rbN[4],  rbN[5],  rbN[6],  rbN[7]);
      const v4bf a2 = cvt4(rbN[8],  rbN[9],  rbN[10], rbN[11]);
      const v4bf a3 = cvt4(rbN[12], rbN[13], rbN[14], rbN[15]);
      *reinterpret_cast<v8bf*>(&dB[ldsBoffN]) =
          __builtin_shufflevector(a0, a1, 0, 1, 2, 3, 4, 5, 6, 7);
      *reinterpret_cast<v8bf*>(&dB[ldsBoffN + 8]) =
          __builtin_shufflevector(a2, a3, 0, 1, 2, 3, 4, 5, 6, 7);
    }
  };

  const v8f vzero = {0.f, 0.f, 0.f, 0.f, 0.f, 0.f, 0.f, 0.f};
  v8f acc[2][4];
#pragma unroll
  for (int s = 0; s < 2; ++s)
#pragma unroll
    for (int j = 0; j < 4; ++j) acc[s][j] = vzero;

  // ---- pipelined main loop ----
  load_tiles(0);
  store_tiles(sA[0], sB[0]);
  __syncthreads();
  int cur = 0;
  for (int kk = 0; kk < K; kk += 32) {
    const bool more = (kk + 32) < K;
    if (more) load_tiles(kk + 32);  // global loads overlap the WMMA below

    v16bf aF[2], bF[4];
#pragma unroll
    for (int s = 0; s < 2; ++s)
      aF[s] = lds_frag(sA[cur], waveM * 32 + s * 16 + lm, half);
#pragma unroll
    for (int j = 0; j < 4; ++j)
      bF[j] = lds_frag(sB[cur], waveN * 64 + j * 16 + lm, half);
#pragma unroll
    for (int s = 0; s < 2; ++s)
#pragma unroll
      for (int j = 0; j < 4; ++j) acc[s][j] = wmma_bf16(aF[s], bF[j], acc[s][j]);

    if (more) {
      store_tiles(sA[cur ^ 1], sB[cur ^ 1]);  // other buffer: no race with reads
      __syncthreads();
      cur ^= 1;
    }
  }

  // ---- epilogue: C/D layout -> lane holds col lm, VGPR r -> row r + 8*half ----
#pragma unroll
  for (int s = 0; s < 2; ++s) {
#pragma unroll
    for (int j = 0; j < 4; ++j) {
      const int col = tileN + waveN * 64 + j * 16 + lm;
      float bv = 0.f;
      if constexpr (EPI >= 1) bv = bias[col];
      const int rbase = tileM + waveM * 32 + s * 16 + half * 8;
#pragma unroll
      for (int r = 0; r < 8; ++r) {
        float v = acc[s][j][r] + bv;
        if constexpr (EPI == 2)  // exact GELU
          v = 0.5f * v * (1.0f + erff(v * 0.70710678118654752f));
        if constexpr (EPI == 3) v += Res[(size_t)(rbase + r) * N + col];
        C[(size_t)(rbase + r) * N + col] = v;
      }
    }
  }
}

// ---------------------------------------------------------------------------
// Embedding gather: one block per token row.
__global__ __launch_bounds__(256) void embed_kernel(const int* __restrict__ ids,
                                                    const float* __restrict__ emb,
                                                    float* __restrict__ x, int Hd) {
  const int row = blockIdx.x;
  const int id  = ids[row];
  const float* src = emb + (size_t)id * Hd;
  float* dst = x + (size_t)row * Hd;
  for (int j = threadIdx.x; j < Hd; j += 256) dst[j] = src[j];
}

// LayerNorm (biased variance, eps=1e-5): one block per row.
__global__ __launch_bounds__(256) void ln_kernel(const float* __restrict__ x,
                                                 const float* __restrict__ g,
                                                 const float* __restrict__ b,
                                                 float* __restrict__ out, int Hd) {
  __shared__ float red[256];
  const int row = blockIdx.x;
  const int t   = threadIdx.x;
  const float* xr = x + (size_t)row * Hd;
  float s = 0.f;
  for (int j = t; j < Hd; j += 256) s += xr[j];
  red[t] = s; __syncthreads();
  for (int o = 128; o > 0; o >>= 1) { if (t < o) red[t] += red[t + o]; __syncthreads(); }
  const float mu = red[0] / (float)Hd;
  __syncthreads();
  float v = 0.f;
  for (int j = t; j < Hd; j += 256) { const float d = xr[j] - mu; v += d * d; }
  red[t] = v; __syncthreads();
  for (int o = 128; o > 0; o >>= 1) { if (t < o) red[t] += red[t + o]; __syncthreads(); }
  const float inv = rsqrtf(red[0] / (float)Hd + 1e-5f);
  float* orow = out + (size_t)row * Hd;
  for (int j = t; j < Hd; j += 256) orow[j] = (xr[j] - mu) * inv * g[j] + b[j];
}

// Row softmax with pre-scale: softmax(scale * x). One block per row.
__global__ __launch_bounds__(256) void softmax_kernel(float* __restrict__ S_,
                                                      int rowLen, float scale) {
  __shared__ float red[256];
  const int t = threadIdx.x;
  float* p = S_ + (size_t)blockIdx.x * rowLen;
  float mx = -3.402823466e38f;
  for (int j = t; j < rowLen; j += 256) mx = fmaxf(mx, p[j]);
  red[t] = mx; __syncthreads();
  for (int o = 128; o > 0; o >>= 1) { if (t < o) red[t] = fmaxf(red[t], red[t + o]); __syncthreads(); }
  mx = red[0]; __syncthreads();
  float sum = 0.f;
  for (int j = t; j < rowLen; j += 256) {
    const float e = expf(scale * (p[j] - mx));
    p[j] = e; sum += e;
  }
  red[t] = sum; __syncthreads();
  for (int o = 128; o > 0; o >>= 1) { if (t < o) red[t] += red[t + o]; __syncthreads(); }
  const float inv = 1.f / red[0];
  for (int j = t; j < rowLen; j += 256) p[j] *= inv;
}

// ---------------------------------------------------------------------------
extern "C" void kernel_launch(void* const* d_in, const int* in_sizes, int n_in,
                              void* d_out, int out_size, void* d_ws, size_t ws_size,
                              hipStream_t stream) {
  (void)in_sizes; (void)n_in; (void)out_size; (void)ws_size;
  constexpr int Hd = 768, FFd = 3072, Ld = 6, Vd = 32000, Bd = 2, Sd = 1024;
  constexpr int Mt = Bd * Sd;  // 2048 token rows
  const float kScale = 0.03608439182435161f;  // 1/sqrt(768)

  const int*   ids  = (const int*)d_in[0];
  const float* emb  = (const float*)d_in[1];
  const float* Wq   = (const float*)d_in[2];
  const float* bq   = (const float*)d_in[3];
  const float* Wk   = (const float*)d_in[4];
  const float* bk   = (const float*)d_in[5];
  const float* Wv   = (const float*)d_in[6];
  const float* bv   = (const float*)d_in[7];
  const float* Wo   = (const float*)d_in[8];
  const float* bo   = (const float*)d_in[9];
  const float* Wup  = (const float*)d_in[10];
  const float* bup  = (const float*)d_in[11];
  const float* Wdn  = (const float*)d_in[12];
  const float* bdn  = (const float*)d_in[13];
  const float* g1   = (const float*)d_in[14];
  const float* b1v  = (const float*)d_in[15];
  const float* g2   = (const float*)d_in[16];
  const float* b2v  = (const float*)d_in[17];
  const float* Wout = (const float*)d_in[18];
  const float* bout = (const float*)d_in[19];
  float* out = (float*)d_out;

  // workspace carve-up (all f32)
  float* ws  = (float*)d_ws;
  size_t off = 0;
  float* x   = ws + off; off += (size_t)Mt * Hd;
  float* h   = ws + off; off += (size_t)Mt * Hd;
  float* qb  = ws + off; off += (size_t)Mt * Hd;
  float* kb  = ws + off; off += (size_t)Mt * Hd;
  float* vb  = ws + off; off += (size_t)Mt * Hd;
  float* ao  = ws + off; off += (size_t)Mt * Hd;
  float* ffn = ws + off; off += (size_t)Mt * FFd;
  float* sc  = ws + off; off += (size_t)Bd * Sd * Sd;

  const long long sQH = (long long)Sd * Hd;  // per-batch stride in q/k/v
  const long long sSS = (long long)Sd * Sd;  // per-batch stride in scores

  embed_kernel<<<Mt, 256, 0, stream>>>(ids, emb, x, Hd);

  for (int l = 0; l < Ld; ++l) {
    const float* Wq_l  = Wq  + (size_t)l * Hd * Hd;
    const float* Wk_l  = Wk  + (size_t)l * Hd * Hd;
    const float* Wv_l  = Wv  + (size_t)l * Hd * Hd;
    const float* Wo_l  = Wo  + (size_t)l * Hd * Hd;
    const float* Wup_l = Wup + (size_t)l * Hd * FFd;
    const float* Wdn_l = Wdn + (size_t)l * FFd * Hd;

    // ---- attention block ----
    ln_kernel<<<Mt, 256, 0, stream>>>(x, g1 + (size_t)l * Hd, b1v + (size_t)l * Hd, h, Hd);

    gemm_bf16_wmma<1, false><<<dim3(Hd / 128, Mt / 128, 1), 256, 0, stream>>>(
        h, Wq_l, bq + (size_t)l * Hd, nullptr, qb, Mt, Hd, Hd, 0, 0, 0);
    gemm_bf16_wmma<1, false><<<dim3(Hd / 128, Mt / 128, 1), 256, 0, stream>>>(
        h, Wk_l, bk + (size_t)l * Hd, nullptr, kb, Mt, Hd, Hd, 0, 0, 0);
    gemm_bf16_wmma<1, false><<<dim3(Hd / 128, Mt / 128, 1), 256, 0, stream>>>(
        h, Wv_l, bv + (size_t)l * Hd, nullptr, vb, Mt, Hd, Hd, 0, 0, 0);

    // scores = q @ k^T   (per batch via grid.z)
    gemm_bf16_wmma<0, true><<<dim3(Sd / 128, Sd / 128, Bd), 256, 0, stream>>>(
        qb, kb, nullptr, nullptr, sc, Sd, Sd, Hd, sQH, sQH, sSS);

    softmax_kernel<<<Bd * Sd, 256, 0, stream>>>(sc, Sd, kScale);

    // attn_out = probs @ v   (per batch)
    gemm_bf16_wmma<0, false><<<dim3(Hd / 128, Sd / 128, Bd), 256, 0, stream>>>(
        sc, vb, nullptr, nullptr, ao, Sd, Hd, Sd, sSS, sQH, sQH);

    // x = x + attn_out @ Wo + bo
    gemm_bf16_wmma<3, false><<<dim3(Hd / 128, Mt / 128, 1), 256, 0, stream>>>(
        ao, Wo_l, bo + (size_t)l * Hd, x, x, Mt, Hd, Hd, 0, 0, 0);

    // ---- FFN block ----
    ln_kernel<<<Mt, 256, 0, stream>>>(x, g2 + (size_t)l * Hd, b2v + (size_t)l * Hd, h, Hd);

    gemm_bf16_wmma<2, false><<<dim3(FFd / 128, Mt / 128, 1), 256, 0, stream>>>(
        h, Wup_l, bup + (size_t)l * FFd, nullptr, ffn, Mt, FFd, Hd, 0, 0, 0);

    gemm_bf16_wmma<3, false><<<dim3(Hd / 128, Mt / 128, 1), 256, 0, stream>>>(
        ffn, Wdn_l, bdn + (size_t)l * Hd, x, x, Mt, Hd, FFd, 0, 0, 0);
  }

  // logits = x @ Wout + bout
  gemm_bf16_wmma<1, false><<<dim3(Vd / 128, Mt / 128, 1), 256, 0, stream>>>(
      x, Wout, bout, nullptr, out, Mt, Vd, Hd, 0, 0, 0);
}
